// WaveletSwinTransformerBlock_82463372083452
// MI455X (gfx1250) — compile-verified, matrix operand order checked by reading, not run on hardware
//
#include <hip/hip_runtime.h>
#include <hip/hip_bf16.h>

// ---------------------------------------------------------------------------
// WaveletSwinTransformerBlock for MI455X (gfx1250, wave32, WMMA f16)
// GEMMs use v_wmma_f32_16x16x32_f16; all f16 tile staging uses the CDNA5
// async path: global_load_async_to_lds_b128 + s_wait_asynccnt.
// Tokens stored window-partitioned: tok = wi*64 + (i*8+j), wi=((b*16+wh)*16+ww)
// ---------------------------------------------------------------------------

typedef _Float16 h16;
typedef __attribute__((ext_vector_type(16))) _Float16 v16h;
typedef __attribute__((ext_vector_type(8)))  float    v8f;

__device__ __forceinline__ void zero8(v8f &a) {
#pragma unroll
  for (int i = 0; i < 8; ++i) a[i] = 0.f;
}

__device__ __forceinline__ v8f wmma16(v16h a, v16h b, v8f c) {
  return __builtin_amdgcn_wmma_f32_16x16x32_f16(false, a, false, b,
                                                (short)0, c, false, false);
}

// low 32 bits of a generic LDS pointer == wave-relative LDS byte offset
__device__ __forceinline__ unsigned lds_off(const void* p) {
  return (unsigned)(uintptr_t)p;
}

// CDNA5 async copy: 16 bytes global -> LDS, tracked by ASYNCcnt
__device__ __forceinline__ void async_cp_b128(unsigned loff, const void* g) {
  asm volatile("global_load_async_to_lds_b128 %0, %1, off"
               :: "v"(loff), "v"((unsigned long long)(uintptr_t)g)
               : "memory");
}
__device__ __forceinline__ void wait_async() {
  asm volatile("s_wait_asynccnt 0" ::: "memory");
}

// A-style fragment (16x32 f16, ISA 7.12.2): lane L holds row row0+(L&15),
// K offsets {0..7,16..23} (L<16) or {8..15,24..31} (L>=16) relative to k0.
__device__ __forceinline__ v16h frag_row(const h16* __restrict__ p, int stride,
                                         int row0, int k0, int lane) {
  const h16* base = p + (size_t)(row0 + (lane & 15)) * stride + k0 + ((lane >> 4) & 1) * 8;
  v16h r;
#pragma unroll
  for (int i = 0; i < 8; ++i) { r[i] = base[i]; r[i + 8] = base[i + 16]; }
  return r;
}

// B fragment gathered from a row-major [K][N] matrix (column-fixed per lane).
__device__ __forceinline__ v16h frag_col(const h16* __restrict__ p, int stride,
                                         int col0, int k0, int lane) {
  const h16* base = p + (size_t)(k0 + ((lane >> 4) & 1) * 8) * stride + col0 + (lane & 15);
  v16h r;
#pragma unroll
  for (int i = 0; i < 8; ++i) {
    r[i]     = base[(size_t)i * stride];
    r[i + 8] = base[(size_t)(i + 16) * stride];
  }
  return r;
}

// ---------------------------------------------------------------------------
// Weight pre-conversion f32 -> f16 (run once per launch; tiny)
// ---------------------------------------------------------------------------
__global__ __launch_bounds__(256) void cvt_f32f16(const float* __restrict__ s,
                                                  h16* __restrict__ d, int n) {
  int i = blockIdx.x * 256 + threadIdx.x;
  if (i < n) d[i] = (h16)s[i];
}

// ---------------------------------------------------------------------------
// Kernel 1: 1x1 convs (concat(LH,HL)->x, HH->z0, Spat->y) per 64-pixel window
// ---------------------------------------------------------------------------
__global__ __launch_bounds__(256) void conv_win_kernel(
    const float* __restrict__ LH, const float* __restrict__ HL,
    const float* __restrict__ HH, const float* __restrict__ Spat,
    const h16* __restrict__ wf_hllh, const float* __restrict__ b_hllh,
    const h16* __restrict__ wf_hh,   const float* __restrict__ b_hh,
    const h16* __restrict__ wf_spat, const float* __restrict__ b_spat,
    h16* __restrict__ xo, h16* __restrict__ yo, float* __restrict__ z0o) {
  const int Wd = 128, HW = 128 * 128;
  __shared__ h16 Fin[64 * 264];   // [pixel][256 ch] : LH|HL|HH|Spat
  __shared__ h16 Wl[256 * 40];    // weight K-slab [256 out][32 k]

  const int wi = blockIdx.x;
  const int b  = wi >> 8;
  const int wh = (wi >> 4) & 15, ww = wi & 15;
  const int h0 = wh * 8, w0 = ww * 8;
  const int tid = threadIdx.x;

  // inputs are f32 -> convert while staging (can't raw-copy)
  const float* srcs[4] = {LH, HL, HH, Spat};
#pragma unroll
  for (int t = 0; t < 4; ++t) {
    const float* s = srcs[t] + (size_t)b * 64 * HW;
    for (int e = tid; e < 64 * 64; e += 256) {
      int c = e >> 6, p = e & 63;
      int i = p >> 3, j = p & 7;
      Fin[p * 264 + t * 64 + c] = (h16)s[(size_t)c * HW + (h0 + i) * Wd + (w0 + j)];
    }
  }

  const unsigned wlb = lds_off(Wl);
  const int lane = tid & 31, wv = tid >> 5;
  const int m0 = (wv & 3) * 16, n0 = (wv >> 2) * 128;

  v8f accX[8], accZ[8], accY[8];
#pragma unroll
  for (int j = 0; j < 8; ++j) { zero8(accX[j]); zero8(accZ[j]); zero8(accY[j]); }

  // Phase X: K=128 over Fin cols [0,128)
  for (int ks = 0; ks < 4; ++ks) {
    __syncthreads();
    for (int e = tid; e < 256 * 4; e += 256) {      // 256 rows x 64B slab
      int n = e >> 2, part = e & 3;
      async_cp_b128(wlb + (unsigned)(n * 40 + part * 8) * 2,
                    wf_hllh + (size_t)n * 128 + ks * 32 + part * 8);
    }
    wait_async();
    __syncthreads();
    v16h a = frag_row(Fin, 264, m0, ks * 32, lane);
#pragma unroll
    for (int j = 0; j < 8; ++j)
      accX[j] = wmma16(a, frag_row(Wl, 40, n0 + j * 16, 0, lane), accX[j]);
  }
  // Phase Z: K=64 over Fin cols [128,192)
  for (int ks = 0; ks < 2; ++ks) {
    __syncthreads();
    for (int e = tid; e < 256 * 4; e += 256) {
      int n = e >> 2, part = e & 3;
      async_cp_b128(wlb + (unsigned)(n * 40 + part * 8) * 2,
                    wf_hh + (size_t)n * 64 + ks * 32 + part * 8);
    }
    wait_async();
    __syncthreads();
    v16h a = frag_row(Fin, 264, m0, 128 + ks * 32, lane);
#pragma unroll
    for (int j = 0; j < 8; ++j)
      accZ[j] = wmma16(a, frag_row(Wl, 40, n0 + j * 16, 0, lane), accZ[j]);
  }
  // Phase Y: K=64 over Fin cols [192,256)
  for (int ks = 0; ks < 2; ++ks) {
    __syncthreads();
    for (int e = tid; e < 256 * 4; e += 256) {
      int n = e >> 2, part = e & 3;
      async_cp_b128(wlb + (unsigned)(n * 40 + part * 8) * 2,
                    wf_spat + (size_t)n * 64 + ks * 32 + part * 8);
    }
    wait_async();
    __syncthreads();
    v16h a = frag_row(Fin, 264, m0, 192 + ks * 32, lane);
#pragma unroll
    for (int j = 0; j < 8; ++j)
      accY[j] = wmma16(a, frag_row(Wl, 40, n0 + j * 16, 0, lane), accY[j]);
  }

  const size_t tok0 = (size_t)wi * 64;
#pragma unroll
  for (int j = 0; j < 8; ++j) {
    int n = n0 + j * 16 + (lane & 15);
    float bx = b_hllh[n], bz = b_hh[n], by = b_spat[n];
#pragma unroll
    for (int r = 0; r < 8; ++r) {
      int m = m0 + r + ((lane >> 4) & 1) * 8;
      size_t off = (tok0 + m) * 256 + n;
      xo[off]  = (h16)(accX[j][r] + bx);
      yo[off]  = (h16)(accY[j][r] + by);
      z0o[off] = accZ[j][r] + bz;
    }
  }
}

// ---------------------------------------------------------------------------
// LayerNorm kernels (wave per token, 256 features, eps=1e-5, biased var)
// ---------------------------------------------------------------------------
__global__ __launch_bounds__(256) void ln_f16_inplace(
    h16* __restrict__ x, const float* __restrict__ g, const float* __restrict__ bt) {
  int t = blockIdx.x * 8 + (threadIdx.x >> 5);
  int lane = threadIdx.x & 31;
  h16* row = x + (size_t)t * 256;
  float v[8], s = 0.f, s2 = 0.f;
#pragma unroll
  for (int k = 0; k < 8; ++k) {
    v[k] = (float)row[k * 32 + lane];
    s += v[k]; s2 += v[k] * v[k];
  }
#pragma unroll
  for (int m = 1; m < 32; m <<= 1) { s += __shfl_xor(s, m, 32); s2 += __shfl_xor(s2, m, 32); }
  float mean = s * (1.f / 256.f);
  float var  = s2 * (1.f / 256.f) - mean * mean;
  float inv  = rsqrtf(var + 1e-5f);
#pragma unroll
  for (int k = 0; k < 8; ++k) {
    int c = k * 32 + lane;
    row[c] = (h16)((v[k] - mean) * inv * g[c] + bt[c]);
  }
}

__global__ __launch_bounds__(256) void ln_f32f16(
    const float* __restrict__ x, const float* __restrict__ g,
    const float* __restrict__ bt, h16* __restrict__ o) {
  int t = blockIdx.x * 8 + (threadIdx.x >> 5);
  int lane = threadIdx.x & 31;
  const float* row = x + (size_t)t * 256;
  float v[8], s = 0.f, s2 = 0.f;
#pragma unroll
  for (int k = 0; k < 8; ++k) {
    v[k] = row[k * 32 + lane];
    s += v[k]; s2 += v[k] * v[k];
  }
#pragma unroll
  for (int m = 1; m < 32; m <<= 1) { s += __shfl_xor(s, m, 32); s2 += __shfl_xor(s2, m, 32); }
  float mean = s * (1.f / 256.f);
  float var  = s2 * (1.f / 256.f) - mean * mean;
  float inv  = rsqrtf(var + 1e-5f);
#pragma unroll
  for (int k = 0; k < 8; ++k) {
    int c = k * 32 + lane;
    o[(size_t)t * 256 + c] = (h16)((v[k] - mean) * inv * g[c] + bt[c]);
  }
}

// ---------------------------------------------------------------------------
// Generic 64x256x256 GEMM: out[t][n] = (sum_k A[t][k]*W[n][k] + bias[n])*scale
// A and W both f16; all staging async-to-LDS.
// ---------------------------------------------------------------------------
__global__ __launch_bounds__(256) void gemm_tok256(
    const h16* __restrict__ A, const h16* __restrict__ Wg,
    const float* __restrict__ bias, h16* __restrict__ out, float scale) {
  __shared__ h16 Al[64 * 264];
  __shared__ h16 Wl[256 * 40];
  const int tid = threadIdx.x;
  const size_t tok0 = (size_t)blockIdx.x * 64;
  const unsigned alb = lds_off(Al), wlb = lds_off(Wl);

  // async stage A tile (64 x 256 f16 = 32KB)
  for (int e = tid; e < 64 * 32; e += 256) {
    int p = e >> 5, c16 = e & 31;
    async_cp_b128(alb + (unsigned)(p * 264 + c16 * 8) * 2,
                  A + (tok0 + p) * 256 + c16 * 8);
  }
  const int lane = tid & 31, wv = tid >> 5;
  const int m0 = (wv & 3) * 16, n0 = (wv >> 2) * 128;
  v8f acc[8];
#pragma unroll
  for (int j = 0; j < 8; ++j) zero8(acc[j]);
  for (int ks = 0; ks < 8; ++ks) {
    __syncthreads();
    for (int e = tid; e < 256 * 4; e += 256) {
      int n = e >> 2, part = e & 3;
      async_cp_b128(wlb + (unsigned)(n * 40 + part * 8) * 2,
                    Wg + (size_t)n * 256 + ks * 32 + part * 8);
    }
    if (ks < 7) __builtin_prefetch(&Wg[(size_t)tid * 256 + (ks + 1) * 32], 0, 1);
    wait_async();
    __syncthreads();
    v16h a = frag_row(Al, 264, m0, ks * 32, lane);
#pragma unroll
    for (int j = 0; j < 8; ++j)
      acc[j] = wmma16(a, frag_row(Wl, 40, n0 + j * 16, 0, lane), acc[j]);
  }
#pragma unroll
  for (int j = 0; j < 8; ++j) {
    int n = n0 + j * 16 + (lane & 15);
    float bb = bias[n];
#pragma unroll
    for (int r = 0; r < 8; ++r) {
      int m = m0 + r + ((lane >> 4) & 1) * 8;
      out[(tok0 + m) * 256 + n] = (h16)((acc[j][r] + bb) * scale);
    }
  }
}

// Projection + residual variant (f32 output zt = A@W^T + b + z0)
__global__ __launch_bounds__(256) void gemm_proj_kernel(
    const h16* __restrict__ A, const h16* __restrict__ Wg,
    const float* __restrict__ bias, const float* __restrict__ resid,
    float* __restrict__ out) {
  __shared__ h16 Al[64 * 264];
  __shared__ h16 Wl[256 * 40];
  const int tid = threadIdx.x;
  const size_t tok0 = (size_t)blockIdx.x * 64;
  const unsigned alb = lds_off(Al), wlb = lds_off(Wl);
  for (int e = tid; e < 64 * 32; e += 256) {
    int p = e >> 5, c16 = e & 31;
    async_cp_b128(alb + (unsigned)(p * 264 + c16 * 8) * 2,
                  A + (tok0 + p) * 256 + c16 * 8);
  }
  const int lane = tid & 31, wv = tid >> 5;
  const int m0 = (wv & 3) * 16, n0 = (wv >> 2) * 128;
  v8f acc[8];
#pragma unroll
  for (int j = 0; j < 8; ++j) zero8(acc[j]);
  for (int ks = 0; ks < 8; ++ks) {
    __syncthreads();
    for (int e = tid; e < 256 * 4; e += 256) {
      int n = e >> 2, part = e & 3;
      async_cp_b128(wlb + (unsigned)(n * 40 + part * 8) * 2,
                    Wg + (size_t)n * 256 + ks * 32 + part * 8);
    }
    if (ks < 7) __builtin_prefetch(&Wg[(size_t)tid * 256 + (ks + 1) * 32], 0, 1);
    wait_async();
    __syncthreads();
    v16h a = frag_row(Al, 264, m0, ks * 32, lane);
#pragma unroll
    for (int j = 0; j < 8; ++j)
      acc[j] = wmma16(a, frag_row(Wl, 40, n0 + j * 16, 0, lane), acc[j]);
  }
#pragma unroll
  for (int j = 0; j < 8; ++j) {
    int n = n0 + j * 16 + (lane & 15);
    float bb = bias[n];
#pragma unroll
    for (int r = 0; r < 8; ++r) {
      int m = m0 + r + ((lane >> 4) & 1) * 8;
      size_t off = (tok0 + m) * 256 + n;
      out[off] = acc[j][r] + bb + resid[off];
    }
  }
}

// ---------------------------------------------------------------------------
// Windowed attention: one wave per (window, head); 64 tokens, hd=32, nH=8.
// Q/K/V head tiles async-staged to LDS (wave-local, no barrier needed).
// ---------------------------------------------------------------------------
__global__ __launch_bounds__(64) void attn_kernel(
    const h16* __restrict__ q, const h16* __restrict__ k,
    const h16* __restrict__ v, const float* __restrict__ rpb,
    h16* __restrict__ o) {
  __shared__ h16 Ql[2 * 64 * 40];
  __shared__ h16 Kl[2 * 64 * 40];
  __shared__ h16 Vl[2 * 64 * 40];
  __shared__ h16 Pl[2 * 64 * 72];
  const int wi = blockIdx.x >> 2;
  const int hp = blockIdx.x & 3;
  const int wv = threadIdx.x >> 5, lane = threadIdx.x & 31;
  const int h  = hp * 2 + wv;
  const size_t base = (size_t)wi * 64 * 256 + (size_t)h * 32;

  h16* Qw = Ql + wv * 64 * 40;
  h16* Kw = Kl + wv * 64 * 40;
  h16* Vw = Vl + wv * 64 * 40;
  const unsigned qo = lds_off(Qw), ko = lds_off(Kw), vo = lds_off(Vw);
  for (int e = lane; e < 64 * 4; e += 32) {   // 64 rows x 64B per tensor
    int row = e >> 2, part = e & 3;
    unsigned l = (unsigned)(row * 40 + part * 8) * 2;
    size_t  g  = base + (size_t)row * 256 + part * 8;
    async_cp_b128(qo + l, q + g);
    async_cp_b128(ko + l, k + g);
    async_cp_b128(vo + l, v + g);
  }
  wait_async();   // wave-local staging: own ASYNCcnt only, no barrier

  v8f S[16];
#pragma unroll
  for (int f = 0; f < 16; ++f) zero8(S[f]);

  v16h af[4];
#pragma unroll
  for (int mi = 0; mi < 4; ++mi) af[mi] = frag_row(Qw, 40, mi * 16, 0, lane);
#pragma unroll
  for (int nj = 0; nj < 4; ++nj) {
    v16h bf = frag_row(Kw, 40, nj * 16, 0, lane);  // rows of K == K^T columns
#pragma unroll
    for (int mi = 0; mi < 4; ++mi)
      S[mi * 4 + nj] = wmma16(af[mi], bf, S[mi * 4 + nj]);
  }

  const int cl = lane & 15, hb = (lane >> 4) & 1;
  // relative position bias: idx = (iq-ik+7)*15 + (jq-jk+7), rpb[idx][h]
#pragma unroll
  for (int nj = 0; nj < 4; ++nj) {
    int c = nj * 16 + cl, i2 = c >> 3, j2 = c & 7;
#pragma unroll
    for (int mi = 0; mi < 4; ++mi)
#pragma unroll
      for (int r = 0; r < 8; ++r) {
        int m = mi * 16 + r + hb * 8;
        int idx = ((m >> 3) - i2 + 7) * 15 + ((m & 7) - j2 + 7);
        S[mi * 4 + nj][r] += rpb[idx * 8 + h];
      }
  }
  // row softmax: each row lives across one 16-lane half (4 frags per lane)
#pragma unroll
  for (int mi = 0; mi < 4; ++mi)
#pragma unroll
    for (int r = 0; r < 8; ++r) {
      float mx = fmaxf(fmaxf(S[mi * 4 + 0][r], S[mi * 4 + 1][r]),
                       fmaxf(S[mi * 4 + 2][r], S[mi * 4 + 3][r]));
      for (int msk = 1; msk < 16; msk <<= 1) mx = fmaxf(mx, __shfl_xor(mx, msk, 32));
      float sum = 0.f;
#pragma unroll
      for (int nj = 0; nj < 4; ++nj) {
        float e = __expf(S[mi * 4 + nj][r] - mx);
        S[mi * 4 + nj][r] = e; sum += e;
      }
      for (int msk = 1; msk < 16; msk <<= 1) sum += __shfl_xor(sum, msk, 32);
      float inv = 1.f / sum;
#pragma unroll
      for (int nj = 0; nj < 4; ++nj) S[mi * 4 + nj][r] *= inv;
    }
  // stage P (f16) per-wave in LDS, then O = P @ V (K=64, N=32)
  h16* P = Pl + wv * 64 * 72;
#pragma unroll
  for (int nj = 0; nj < 4; ++nj) {
    int c = nj * 16 + cl;
#pragma unroll
    for (int mi = 0; mi < 4; ++mi)
#pragma unroll
      for (int r = 0; r < 8; ++r)
        P[(mi * 16 + r + hb * 8) * 72 + c] = (h16)S[mi * 4 + nj][r];
  }
  v8f O[8];
#pragma unroll
  for (int f = 0; f < 8; ++f) zero8(O[f]);
#pragma unroll
  for (int kb2 = 0; kb2 < 2; ++kb2) {
#pragma unroll
    for (int nj = 0; nj < 2; ++nj) {
      v16h bf = frag_col(Vw, 40, nj * 16, kb2 * 32, lane);
#pragma unroll
      for (int mi = 0; mi < 4; ++mi) {
        v16h a = frag_row(P, 72, mi * 16, kb2 * 32, lane);
        O[mi * 2 + nj] = wmma16(a, bf, O[mi * 2 + nj]);
      }
    }
  }
#pragma unroll
  for (int mi = 0; mi < 4; ++mi)
#pragma unroll
    for (int nj = 0; nj < 2; ++nj) {
      int n = nj * 16 + cl;
#pragma unroll
      for (int r = 0; r < 8; ++r) {
        int m = mi * 16 + r + hb * 8;
        o[((size_t)wi * 64 + m) * 256 + h * 32 + n] = (h16)O[mi * 2 + nj][r];
      }
    }
}

// ---------------------------------------------------------------------------
// Fused MLP: out = zt + fc2(gelu(fc1(ztn))) -> written directly as NCHW.
// Hidden (1024) processed in 16 chunks of 64; weight slabs async-staged.
// ---------------------------------------------------------------------------
__global__ __launch_bounds__(256) void mlp_kernel(
    const h16* __restrict__ ztn, const float* __restrict__ zt,
    const h16* __restrict__ fc1w, const float* __restrict__ fc1b,
    const h16* __restrict__ fc2w, const float* __restrict__ fc2b,
    float* __restrict__ out) {
  __shared__ h16 Al[64 * 264];   // ztn tile
  __shared__ h16 Hl[64 * 72];    // gelu(hidden chunk) 64x64
  __shared__ h16 Wu[256 * 40];   // shared weight slab (fc1 & fc2 alternate)
  const int tid = threadIdx.x;
  const size_t tok0 = (size_t)blockIdx.x * 64;
  const unsigned alb = lds_off(Al), wub = lds_off(Wu);
  for (int e = tid; e < 64 * 32; e += 256) {
    int p = e >> 5, c16 = e & 31;
    async_cp_b128(alb + (unsigned)(p * 264 + c16 * 8) * 2,
                  ztn + (tok0 + p) * 256 + c16 * 8);
  }
  const int lane = tid & 31, wv = tid >> 5;
  const int m0 = (wv & 3) * 16, n0 = (wv >> 2) * 128;  // out tile mapping
  const int n0h = (wv >> 2) * 32;                      // hidden tile mapping
  v8f accO[8];
#pragma unroll
  for (int j = 0; j < 8; ++j) zero8(accO[j]);

  for (int hc = 0; hc < 16; ++hc) {
    v8f accH[2];
    zero8(accH[0]); zero8(accH[1]);
    // hchunk = ztn_tile @ fc1_w[hc*64:+64, :]^T   (K=256)
    for (int ks = 0; ks < 8; ++ks) {
      __syncthreads();
      for (int e = tid; e < 64 * 4; e += 256) {     // 64 rows x 64B slab
        int n = e >> 2, part = e & 3;
        async_cp_b128(wub + (unsigned)(n * 40 + part * 8) * 2,
                      fc1w + (size_t)(hc * 64 + n) * 256 + ks * 32 + part * 8);
      }
      wait_async();
      __syncthreads();
      v16h a = frag_row(Al, 264, m0, ks * 32, lane);
#pragma unroll
      for (int j = 0; j < 2; ++j)
        accH[j] = wmma16(a, frag_row(Wu, 40, n0h + j * 16, 0, lane), accH[j]);
    }
    __syncthreads();
    // bias + exact GELU -> LDS
#pragma unroll
    for (int j = 0; j < 2; ++j) {
      int n = n0h + j * 16 + (lane & 15);
      float bb = fc1b[hc * 64 + n];
#pragma unroll
      for (int r = 0; r < 8; ++r) {
        int m = m0 + r + ((lane >> 4) & 1) * 8;
        float xv = accH[j][r] + bb;
        Hl[m * 72 + n] = (h16)(0.5f * xv * (1.0f + erff(xv * 0.7071067811865475f)));
      }
    }
    __syncthreads();
    // accO += Hchunk @ fc2_w[:, hc*64:+64]^T   (K=64)
    for (int ks = 0; ks < 2; ++ks) {
      __syncthreads();
      for (int e = tid; e < 256 * 4; e += 256) {
        int n = e >> 2, part = e & 3;
        async_cp_b128(wub + (unsigned)(n * 40 + part * 8) * 2,
                      fc2w + (size_t)n * 1024 + hc * 64 + ks * 32 + part * 8);
      }
      wait_async();
      __syncthreads();
      v16h a = frag_row(Hl, 72, m0, ks * 32, lane);
#pragma unroll
      for (int j = 0; j < 8; ++j)
        accO[j] = wmma16(a, frag_row(Wu, 40, n0 + j * 16, 0, lane), accO[j]);
    }
    __syncthreads();
  }

  // epilogue: + fc2_b + zt residual, window-reverse, NCHW store
  const int wi = blockIdx.x;
  const int b = wi >> 8, wh = (wi >> 4) & 15, ww = wi & 15;
#pragma unroll
  for (int j = 0; j < 8; ++j) {
    int n = n0 + j * 16 + (lane & 15);
    float bb = fc2b[n];
#pragma unroll
    for (int r = 0; r < 8; ++r) {
      int m = m0 + r + ((lane >> 4) & 1) * 8;
      size_t tok = tok0 + m;
      float val = accO[j][r] + bb + zt[tok * 256 + n];
      int hh = wh * 8 + (m >> 3), www = ww * 8 + (m & 7);
      out[(((size_t)b * 256 + n) * 128 + hh) * 128 + www] = val;
    }
  }
}

// ---------------------------------------------------------------------------
// Launch
// ---------------------------------------------------------------------------
extern "C" void kernel_launch(void* const* d_in, const int* in_sizes, int n_in,
                              void* d_out, int out_size, void* d_ws, size_t ws_size,
                              hipStream_t stream) {
  (void)in_sizes; (void)n_in; (void)out_size; (void)ws_size;
  const float* LH     = (const float*)d_in[0];
  const float* HL     = (const float*)d_in[1];
  const float* HH     = (const float*)d_in[2];
  const float* Spat   = (const float*)d_in[3];
  const float* w_hllh = (const float*)d_in[4];
  const float* b_hllh = (const float*)d_in[5];
  const float* w_hh   = (const float*)d_in[6];
  const float* b_hh   = (const float*)d_in[7];
  const float* w_spat = (const float*)d_in[8];
  const float* b_spat = (const float*)d_in[9];
  const float* ln1x_g = (const float*)d_in[10];
  const float* ln1x_b = (const float*)d_in[11];
  const float* ln1y_g = (const float*)d_in[12];
  const float* ln1y_b = (const float*)d_in[13];
  const float* qkv_w  = (const float*)d_in[14];
  const float* qkv_b  = (const float*)d_in[15];
  const float* qkv2_w = (const float*)d_in[16];
  const float* qkv2_b = (const float*)d_in[17];
  const float* qkv3_w = (const float*)d_in[18];
  const float* qkv3_b = (const float*)d_in[19];
  const float* proj_w = (const float*)d_in[20];
  const float* proj_b = (const float*)d_in[21];
  const float* rpb    = (const float*)d_in[22];
  const float* ln2_g  = (const float*)d_in[23];
  const float* ln2_b  = (const float*)d_in[24];
  const float* fc1w   = (const float*)d_in[25];
  const float* fc1b   = (const float*)d_in[26];
  const float* fc2w   = (const float*)d_in[27];
  const float* fc2b   = (const float*)d_in[28];

  const size_t T = (size_t)8 * 128 * 128;  // 131072 tokens
  char* ws = (char*)d_ws;
  // workspace layout (peak ~472 MB), with region reuse:
  h16*   x   = (h16*)ws;                                  // T*256 f16
  h16*   y   = (h16*)(ws + T * 256 * 2);                  // T*256 f16
  float* z0  = (float*)(ws + 2 * T * 256 * 2);            // T*256 f32
  h16*   q   = (h16*)(ws + 2 * T * 256 * 2 + T * 256 * 4);
  h16*   kk  = q + T * 256;
  h16*   v   = kk + T * 256;
  h16*   attno = x;           // reuse x after QKV consume it
  float* zt  = (float*)q;     // reuse q+k after attention
  h16*   ztn = v;             // reuse v after attention
  // f16 weight cache (~1.7 MB) after q/k/v region
  h16* wf      = v + T * 256;
  h16* c_hllh  = wf;               // 256*128
  h16* c_hh    = c_hllh + 32768;   // 256*64
  h16* c_spat  = c_hh   + 16384;   // 256*64
  h16* c_q     = c_spat + 16384;   // 256*256
  h16* c_k     = c_q    + 65536;
  h16* c_v     = c_k    + 65536;
  h16* c_proj  = c_v    + 65536;
  h16* c_fc1   = c_proj + 65536;   // 1024*256
  h16* c_fc2   = c_fc1  + 262144;  // 256*1024

  // weight pre-conversion (f32 -> f16, once per launch)
  cvt_f32f16<<<(32768 + 255) / 256, 256, 0, stream>>>(w_hllh, c_hllh, 32768);
  cvt_f32f16<<<(16384 + 255) / 256, 256, 0, stream>>>(w_hh,   c_hh,   16384);
  cvt_f32f16<<<(16384 + 255) / 256, 256, 0, stream>>>(w_spat, c_spat, 16384);
  cvt_f32f16<<<(65536 + 255) / 256, 256, 0, stream>>>(qkv_w,  c_q,    65536);
  cvt_f32f16<<<(65536 + 255) / 256, 256, 0, stream>>>(qkv2_w, c_k,    65536);
  cvt_f32f16<<<(65536 + 255) / 256, 256, 0, stream>>>(qkv3_w, c_v,    65536);
  cvt_f32f16<<<(65536 + 255) / 256, 256, 0, stream>>>(proj_w, c_proj, 65536);
  cvt_f32f16<<<(262144 + 255) / 256, 256, 0, stream>>>(fc1w,  c_fc1,  262144);
  cvt_f32f16<<<(262144 + 255) / 256, 256, 0, stream>>>(fc2w,  c_fc2,  262144);

  const int tiles = (int)(T / 64);  // 2048 windows
  conv_win_kernel<<<tiles, 256, 0, stream>>>(LH, HL, HH, Spat,
      c_hllh, b_hllh, c_hh, b_hh, c_spat, b_spat, x, y, z0);
  ln_f16_inplace<<<(int)(T / 8), 256, 0, stream>>>(x, ln1x_g, ln1x_b);
  ln_f16_inplace<<<(int)(T / 8), 256, 0, stream>>>(y, ln1y_g, ln1y_b);
  gemm_tok256<<<tiles, 256, 0, stream>>>(x, c_q, qkv_b,  q,  0.17677669529663689f);
  gemm_tok256<<<tiles, 256, 0, stream>>>(y, c_k, qkv2_b, kk, 1.0f);
  gemm_tok256<<<tiles, 256, 0, stream>>>(y, c_v, qkv3_b, v,  1.0f);
  attn_kernel<<<tiles * 4, 64, 0, stream>>>(q, kk, v, rpb, attno);
  gemm_proj_kernel<<<tiles, 256, 0, stream>>>(attno, c_proj, proj_b, z0, zt);
  ln_f32f16<<<(int)(T / 8), 256, 0, stream>>>(zt, ln2_g, ln2_b, ztn);
  mlp_kernel<<<tiles, 256, 0, stream>>>(ztn, zt, c_fc1, fc1b, c_fc2, fc2b, (float*)d_out);
}
